// MultiStepLinearRNN_16458314678810
// MI455X (gfx1250) — matrix operand order repaired and checked by reading, not compiled, not run on hardware
//
#include <hip/hip_runtime.h>

// MultiStepLinearRNN on gfx1250:
//   B=64, T=512, D_IN=128, D_H=512, F=32
// 4 workgroups (16 batch rows each), 512 threads = 16 waves; each wave owns
// TWO 16-column output tiles (w and w+16) so each LDS A-fragment load feeds
// two v_wmma_f32_16x16x32_bf16 ops. h0/h1 live in LDS as bf16 A-fragments;
// weights pre-packed to bf16 B-fragments in d_ws (L2-resident, ~1.8 MB).

#define TSEQ   512
#define DIN    128
#define DH     512
#define FSTEPS 32

typedef __attribute__((ext_vector_type(16))) __bf16 v16bf;
typedef __attribute__((ext_vector_type(8)))  float  v8f;

__device__ __forceinline__ unsigned short f2bf_bits(float f) {
  __bf16 h = (__bf16)f;
  return __builtin_bit_cast(unsigned short, h);
}

__device__ __forceinline__ v8f wmma_bf16(v16bf a, v16bf b, v8f c) {
  // (neg_a, A, neg_b, B, c_mod, C, reuse_a, reuse_b)
  return __builtin_amdgcn_wmma_f32_16x16x32_bf16(false, a, false, b, (short)0, c,
                                                 false, false);
}

// ---- fragment accessors -----------------------------------------------------
// A fragment (16x32 bf16) in LDS: 1 KB per k-block; per-lane 32B chunk at lane*32.
__device__ __forceinline__ v16bf lds_afrag(const unsigned char* buf, int kb, int lane) {
  return *reinterpret_cast<const v16bf*>(buf + kb * 1024 + lane * 32);
}
// B fragment (32x16 bf16) pre-packed in global: 512 ushorts per fragment.
__device__ __forceinline__ v16bf gbl_bfrag(const unsigned short* wp, int frag, int lane) {
  return *reinterpret_cast<const v16bf*>(wp + (size_t)frag * 512 + lane * 16);
}

// Dual-tile GEMM: acc{0,1} += A(lds) x B(tiles w0,w1), K = nKb*32.
// One A fragment load feeds two WMMAs.
__device__ __forceinline__ void gemm2(const unsigned char* abuf,
                                      const unsigned short* wpack, int nKb,
                                      int w0, int w1, int lane,
                                      v8f& acc0, v8f& acc1) {
  #pragma unroll 2
  for (int kb = 0; kb < nKb; ++kb) {
    v16bf a  = lds_afrag(abuf, kb, lane);
    v16bf b0 = gbl_bfrag(wpack, w0 * nKb + kb, lane);
    v16bf b1 = gbl_bfrag(wpack, w1 * nKb + kb, lane);
    if (kb + 1 < nKb)
      __builtin_prefetch(wpack + (size_t)(w0 * nKb + kb + 1) * 512 + lane * 16, 0, 3);
    acc0 = wmma_bf16(a, b0, acc0);
    acc1 = wmma_bf16(a, b1, acc1);
  }
}

// Single-tile GEMM (used by the output projection, 8 tiles only).
__device__ __forceinline__ v8f gemm1(const unsigned char* abuf,
                                     const unsigned short* wpack,
                                     int nKb, int w, int lane, v8f acc) {
  #pragma unroll 2
  for (int kb = 0; kb < nKb; ++kb) {
    v16bf a = lds_afrag(abuf, kb, lane);
    v16bf b = gbl_bfrag(wpack, w * nKb + kb, lane);
    acc = wmma_bf16(a, b, acc);
  }
  return acc;
}

// Scatter a 16x16 f32 C/D tile (columns 16w..16w+15) into an LDS bf16
// A-fragment buffer (tile's N index becomes K of the next multiply).
// C/D layout: reg r, lanes 0-15: (M=r, N=lane); lanes 16-31: (M=r+8, N=lane-16).
// A layout (16x32 bf16): g=kk>>3; lane=m+16*(g&1); v=4*(g>>1)+((kk&7)>>1); half=kk&1.
__device__ __forceinline__ void store_tile_to_lds(unsigned char* buf, int w, int lane,
                                                  v8f acc) {
  int nloc = lane & 15;
  int mh   = (lane >> 4) * 8;
  int kk   = ((w & 1) << 4) | nloc;          // k % 32
  int g    = kk >> 3;
  int vt   = ((g >> 1) << 2) + ((kk & 7) >> 1);
  int half = kk & 1;
  int base = (w >> 1) * 1024 + vt * 4 + half * 2;
  #pragma unroll
  for (int r = 0; r < 8; ++r) {
    int m      = mh + r;
    int lane_t = m + ((g & 1) << 4);
    *reinterpret_cast<__bf16*>(buf + base + lane_t * 32) =
        __builtin_bit_cast(__bf16, f2bf_bits(acc[r]));
  }
}

// Load one 16x128 f32 x-tile (row stride = strideF floats) into the LDS
// bf16 A-fragment x buffer. 512 threads x 4 consecutive floats each:
// one float4 global load + one 8-byte DS store.
__device__ __forceinline__ void load_x_tile(unsigned char* xbuf, const float* src,
                                            int strideF, int tid) {
  int e = tid * 4;            // 0..2044
  int m = e >> 7;             // batch row in tile
  int k = e & 127;            // feature (multiple of 4)
  float4 v = *reinterpret_cast<const float4*>(src + (size_t)m * strideF + k);
  int kk     = k & 31;        // kk&7 in {0,4} -> all 4 values share group g
  int g      = kk >> 3;
  int vt     = ((g >> 1) << 2) + ((kk & 7) >> 1);  // even
  int lane_t = m + ((g & 1) << 4);
  unsigned int lo = (unsigned int)f2bf_bits(v.x) | ((unsigned int)f2bf_bits(v.y) << 16);
  unsigned int hi = (unsigned int)f2bf_bits(v.z) | ((unsigned int)f2bf_bits(v.w) << 16);
  uint2 pk; pk.x = lo; pk.y = hi;
  *reinterpret_cast<uint2*>(xbuf + (k >> 5) * 1024 + lane_t * 32 + vt * 4) = pk;
}

__device__ __forceinline__ v8f splat8(float f) {
  v8f a;
  #pragma unroll
  for (int r = 0; r < 8; ++r) a[r] = f;
  return a;
}

// One RNN step (dual-tile per wave):
//   h0 = d0*(h0@Wh0.T) + x@Wx0.T + b0 ; h1 = d1*(h1@Wh1.T) + h0@Wx1.T + b1
__device__ void rnn_step(unsigned char* h0b, unsigned char* h1b, unsigned char* xb,
                         const unsigned short* wh0, const unsigned short* wx0,
                         const unsigned short* wh1, const unsigned short* wx1,
                         int w, int lane, float d0, float d1,
                         float bias0a, float bias0b, float bias1a, float bias1b) {
  const int w0 = w, w1 = w + 16;
  // layer 0
  v8f aH0 = splat8(0.f), aH1 = splat8(0.f);
  gemm2(h0b, wh0, DH / 32, w0, w1, lane, aH0, aH1);
  v8f aX0 = splat8(bias0a), aX1 = splat8(bias0b);
  gemm2(xb, wx0, DIN / 32, w0, w1, lane, aX0, aX1);
  v8f h0n0, h0n1;
  #pragma unroll
  for (int r = 0; r < 8; ++r) {
    h0n0[r] = fmaf(d0, aH0[r], aX0[r]);
    h0n1[r] = fmaf(d0, aH1[r], aX1[r]);
  }
  __syncthreads();                       // all reads of h0b / xb done
  store_tile_to_lds(h0b, w0, lane, h0n0);
  store_tile_to_lds(h0b, w1, lane, h0n1);
  __syncthreads();                       // new h0 visible
  // layer 1 (consumes updated h0)
  v8f bH0 = splat8(0.f), bH1 = splat8(0.f);
  gemm2(h1b, wh1, DH / 32, w0, w1, lane, bH0, bH1);
  v8f bX0 = splat8(bias1a), bX1 = splat8(bias1b);
  gemm2(h0b, wx1, DH / 32, w0, w1, lane, bX0, bX1);
  v8f h1n0, h1n1;
  #pragma unroll
  for (int r = 0; r < 8; ++r) {
    h1n0[r] = fmaf(d1, bH0[r], bX0[r]);
    h1n1[r] = fmaf(d1, bH1[r], bX1[r]);
  }
  __syncthreads();                       // all reads of h1b done
  store_tile_to_lds(h1b, w0, lane, h1n0);
  store_tile_to_lds(h1b, w1, lane, h1n1);
  __syncthreads();                       // new h1 visible
}

// ---- weight pack: f32 row-major W[nOut][kIn] -> bf16 B-fragments ------------
// Fragment f = ntile*(kIn/32)+kb, 512 ushorts: lane L holds N=ntile*16+(L&15),
// K = kb*32 + ((L>>4)<<3) + (j&7) + ((j>>3)<<4) for j=0..15.
__global__ void pack_w_kernel(const float* __restrict__ src,
                              unsigned short* __restrict__ dst, int nOut, int kIn) {
  int p = blockIdx.x * blockDim.x + threadIdx.x;
  if (p >= nOut * kIn) return;
  int frag  = p >> 9;
  int rem   = p & 511;
  int lane  = rem >> 4;
  int j     = rem & 15;
  int nKb   = kIn >> 5;
  int ntile = frag / nKb;
  int kb    = frag - ntile * nKb;
  int n = ntile * 16 + (lane & 15);
  int k = kb * 32 + ((lane >> 4) << 3) + (j & 7) + ((j >> 3) << 4);
  dst[p] = f2bf_bits(src[(size_t)n * kIn + k]);
}

// ---- main recurrent kernel --------------------------------------------------
__global__ __launch_bounds__(512, 1)
void rnn_main_kernel(const float* __restrict__ x_in,
                     const float* __restrict__ b0p, const float* __restrict__ d0p,
                     const float* __restrict__ b1p, const float* __restrict__ d1p,
                     const float* __restrict__ bpp,
                     const unsigned short* __restrict__ wh0,
                     const unsigned short* __restrict__ wx0,
                     const unsigned short* __restrict__ wx1,
                     const unsigned short* __restrict__ wh1,
                     const unsigned short* __restrict__ wpp,
                     float* __restrict__ out) {
  __shared__ alignas(32) unsigned char lds_h0[16 * DH * 2];   // 16 KB
  __shared__ alignas(32) unsigned char lds_h1[16 * DH * 2];   // 16 KB
  __shared__ alignas(32) unsigned char lds_x [16 * DIN * 2];  //  4 KB

  const int tid  = threadIdx.x;
  const int lane = tid & 31;
  const int w    = tid >> 5;           // wave id 0..15; owns tiles w and w+16
  const int bt   = blockIdx.x;         // batch tile 0..3 (16 rows each)

  const float d0 = d0p[0];
  const float d1 = d1p[0];
  const int   nloc = lane & 15;
  const float bias0a = b0p[w * 16 + nloc];
  const float bias0b = b0p[(w + 16) * 16 + nloc];
  const float bias1a = b1p[w * 16 + nloc];
  const float bias1b = b1p[(w + 16) * 16 + nloc];
  const float biasp  = (w < 8) ? bpp[w * 16 + nloc] : 0.f;

  // h0 = h1 = 0
  for (int i = tid; i < (16 * DH * 2) / 4; i += 512) {
    reinterpret_cast<unsigned int*>(lds_h0)[i] = 0u;
    reinterpret_cast<unsigned int*>(lds_h1)[i] = 0u;
  }
  __syncthreads();

  // ---- encode: scan over T ----
  const float* xbase = x_in + (size_t)bt * 16 * TSEQ * DIN;
  for (int t = 0; t < TSEQ; ++t) {
    load_x_tile(lds_x, xbase + (size_t)t * DIN, TSEQ * DIN, tid);
    __syncthreads();
    rnn_step(lds_h0, lds_h1, lds_x, wh0, wx0, wh1, wx1,
             w, lane, d0, d1, bias0a, bias0b, bias1a, bias1b);
  }

  // ---- decode: project, emit, feed back, step ----
  for (int f = 0; f < FSTEPS; ++f) {
    if (w < 8) {                          // D_IN/16 = 8 output tiles
      v8f acc = gemm1(lds_h1, wpp, DH / 32, w, lane, splat8(biasp));
      const int mh = (lane >> 4) * 8;
      #pragma unroll
      for (int r = 0; r < 8; ++r) {
        int m = mh + r;
        int n = w * 16 + nloc;
        out[((size_t)(bt * 16 + m) * FSTEPS + f) * DIN + n] = acc[r];
      }
      store_tile_to_lds(lds_x, w, lane, acc);   // pred becomes next x
    }
    __syncthreads();
    if (f + 1 < FSTEPS) {
      rnn_step(lds_h0, lds_h1, lds_x, wh0, wx0, wh1, wx1,
               w, lane, d0, d1, bias0a, bias0b, bias1a, bias1b);
    }
  }
}

extern "C" void kernel_launch(void* const* d_in, const int* in_sizes, int n_in,
                              void* d_out, int out_size, void* d_ws, size_t ws_size,
                              hipStream_t stream) {
  (void)in_sizes; (void)n_in; (void)out_size; (void)ws_size;
  const float* x   = (const float*)d_in[0];
  const float* Wx0 = (const float*)d_in[1];
  const float* b0  = (const float*)d_in[2];
  const float* Wh0 = (const float*)d_in[3];
  const float* d0  = (const float*)d_in[4];
  const float* Wx1 = (const float*)d_in[5];
  const float* b1  = (const float*)d_in[6];
  const float* Wh1 = (const float*)d_in[7];
  const float* d1  = (const float*)d_in[8];
  const float* Wp  = (const float*)d_in[9];
  const float* bp  = (const float*)d_in[10];

  // d_ws layout (ushort units): packed bf16 weights, ~1.75 MB total
  unsigned short* ws   = (unsigned short*)d_ws;
  unsigned short* wh0p = ws;                    // 512*512
  unsigned short* wx0p = ws + 262144;           // 512*128
  unsigned short* wx1p = ws + 327680;           // 512*512
  unsigned short* wh1p = ws + 589824;           // 512*512
  unsigned short* wpp  = ws + 851968;           // 128*512

  pack_w_kernel<<<(512 * 512 + 255) / 256, 256, 0, stream>>>(Wh0, wh0p, 512, 512);
  pack_w_kernel<<<(512 * 128 + 255) / 256, 256, 0, stream>>>(Wx0, wx0p, 512, 128);
  pack_w_kernel<<<(512 * 512 + 255) / 256, 256, 0, stream>>>(Wx1, wx1p, 512, 512);
  pack_w_kernel<<<(512 * 512 + 255) / 256, 256, 0, stream>>>(Wh1, wh1p, 512, 512);
  pack_w_kernel<<<(128 * 512 + 255) / 256, 256, 0, stream>>>(Wp,  wpp,  128, 512);

  rnn_main_kernel<<<4, 512, 0, stream>>>(x, b0, d0, b1, d1, bp,
                                         wh0p, wx0p, wx1p, wh1p, wpp,
                                         (float*)d_out);
}
